// Capsule_7584912245126
// MI455X (gfx1250) — compile-verified, matrix operand order checked by reading, not run on hardware
//
#include <hip/hip_runtime.h>
#include <hip/hip_bf16.h>
#include <cstdint>
#include <cstddef>

// Capsule routing, MI455X (gfx1250, wave32).
// Schedule (all on `stream`, serialized, graph-capturable):
//   K0 zero(S0,s1,s2)
//   K1 gemm_hat: hat[b,n,k*64+m] = (x@W) in bf16 (WMMA bf16, fp32 acc),
//                fused column-sum S0[b,j] = sum_n hat  (atomics)
//   K2 squash:   o_sum = squash(S0/32)                 (= outputs0)
//   K3 route:    logits = o_sum . hat ; softmax_k ; s1 += c*hat
//   K4 squash:   o_sum += squash(s1)                   (= outputs0+outputs1)
//   K5 route:    same with o_sum -> s2
//   K6 squash:   out = squash(s2)
// Workspace: hat 256MB + S0/s1/s2/o_sum (4 x 256KB).

#define B_    32
#define N_    2048
#define D_    256
#define NCAP  32
#define MCAP  64
#define NKM   2048   // NCAP*MCAP

typedef __attribute__((ext_vector_type(16))) __bf16 v16bf;
typedef __attribute__((ext_vector_type(4)))  __bf16 v4bf;
typedef __attribute__((ext_vector_type(8)))  float  v8f;

struct Bits32 { uint4 lo, hi; };  // 32 bytes -> v16bf

__device__ __forceinline__ v16bf frag_from_lds(const __bf16* p0, const __bf16* p1) {
    Bits32 t;
    t.lo = *reinterpret_cast<const uint4*>(p0);  // K {0..7} or {8..15}  (half-dependent)
    t.hi = *reinterpret_cast<const uint4*>(p1);  // K {16..23} or {24..31}
    return __builtin_bit_cast(v16bf, t);
}

__device__ __forceinline__ float bflo(unsigned u) { return __builtin_bit_cast(float, u << 16); }
__device__ __forceinline__ float bfhi(unsigned u) { return __builtin_bit_cast(float, u & 0xffff0000u); }

__device__ __forceinline__ float squash_scale(float ss) {
    float s = ss + 1e-7f;
    return __fsqrt_rn(s) / (0.5f + s);
}

// ---------------------------------------------------------------- K0: zero
__global__ void zero_kernel(float* __restrict__ p, int n) {
    int i = blockIdx.x * 256 + threadIdx.x;
    if (i < n) p[i] = 0.f;
}

// ---------------------------------------------------------------- K1: GEMM
// Grid (16 colTiles, 16 rowTiles, 32 B), 256 threads = 8 waves.
// WG tile 128x128; wave tile 32x64 = 2x4 fragments of 16x16, K unrolled 8x32.
#define AP 264   // LDS row stride in bf16 (256 + 8 pad): row-to-row shift = 4 banks

__global__ __launch_bounds__(256)
void gemm_hat_kernel(const float* __restrict__ x, const float* __restrict__ W,
                     __bf16* __restrict__ hat, float* __restrict__ S0) {
    __shared__ __bf16 As[128 * AP];   // x slab,  [row][k]
    __shared__ __bf16 Bs[128 * AP];   // W slab transposed, [col][k]
    __shared__ float  colsum[128];

    const int tid  = threadIdx.x;
    const int b    = blockIdx.z;
    const int row0 = blockIdx.y * 128;
    const int col0 = blockIdx.x * 128;

    if (tid < 128) colsum[tid] = 0.f;

    // stage x slab (128 rows x 256 K), float4 loads, convert to bf16
    {
        const float4* xg = reinterpret_cast<const float4*>(x + ((size_t)b * N_ + row0) * D_);
        #pragma unroll 4
        for (int i = 0; i < 32; ++i) {
            int f4 = i * 256 + tid;
            int r  = f4 >> 6;             // 64 float4 per row
            int kq = (f4 & 63) << 2;
            float4 v = xg[(size_t)r * 64 + (f4 & 63)];
            v4bf h;
            h.x = (__bf16)v.x; h.y = (__bf16)v.y; h.z = (__bf16)v.z; h.w = (__bf16)v.w;
            *reinterpret_cast<v4bf*>(&As[r * AP + kq]) = h;
        }
    }
    // stage W slab transposed: Bs[col][k]
    {
        const float4* wg = reinterpret_cast<const float4*>(W);
        #pragma unroll 4
        for (int i = 0; i < 32; ++i) {
            int f4 = i * 256 + tid;
            int k  = f4 >> 5;             // 32 float4 per 128-col row
            int cq = (f4 & 31) << 2;
            float4 v = wg[((size_t)k * NKM + col0 + cq) >> 2];
            Bs[(cq + 0) * AP + k] = (__bf16)v.x;
            Bs[(cq + 1) * AP + k] = (__bf16)v.y;
            Bs[(cq + 2) * AP + k] = (__bf16)v.z;
            Bs[(cq + 3) * AP + k] = (__bf16)v.w;
        }
    }
    __syncthreads();

    const int lane = tid & 31;
    const int w    = tid >> 5;
    const int wr   = (w >> 1) * 32;        // wave row base
    const int wc   = (w & 1) * 64;         // wave col base
    const int l16  = lane & 15;
    const int half = lane >> 4;
    const int hoff = half * 8;             // ISA 16-bit A layout: lanes>=16 hold K+8

    v8f acc[2][4] = {};

    #pragma unroll
    for (int kk = 0; kk < 256; kk += 32) {
        v16bf a[2], bm[4];
        #pragma unroll
        for (int i = 0; i < 2; ++i) {
            const __bf16* p = &As[(wr + i * 16 + l16) * AP + kk + hoff];
            a[i] = frag_from_lds(p, p + 16);
        }
        #pragma unroll
        for (int j = 0; j < 4; ++j) {
            const __bf16* p = &Bs[(wc + j * 16 + l16) * AP + kk + hoff];
            bm[j] = frag_from_lds(p, p + 16);
        }
        #pragma unroll
        for (int i = 0; i < 2; ++i)
            #pragma unroll
            for (int j = 0; j < 4; ++j)
                acc[i][j] = __builtin_amdgcn_wmma_f32_16x16x32_bf16(
                    false, a[i], false, bm[j], (short)0, acc[i][j], false, false);
    }

    // epilogue: store hat (bf16), fused per-column partial sums -> S0
    #pragma unroll
    for (int i = 0; i < 2; ++i) {
        #pragma unroll
        for (int j = 0; j < 4; ++j) {
            int colL = wc + j * 16 + l16;
            size_t cbase = ((size_t)b * N_ + row0 + wr + i * 16) * NKM + col0 + colL;
            float psum = 0.f;
            #pragma unroll
            for (int v = 0; v < 8; ++v) {
                float val = acc[i][j][v];
                int rrow = half ? (v + 8) : v;        // ISA C-layout: lanes>=16 hold M+8
                hat[cbase + (size_t)rrow * NKM] = (__bf16)val;
                psum += val;
            }
            atomicAdd(&colsum[colL], psum);
        }
    }
    __syncthreads();
    if (tid < 128) atomicAdd(&S0[(size_t)b * NKM + col0 + tid], colsum[tid]);
}

// ---------------------------------------------------------------- K3/K5: routing pass
// Grid (16 nChunks, 32 B), 256 threads = 8 waves; lane <-> capsule k.
__global__ __launch_bounds__(256)
void route_kernel(const __bf16* __restrict__ hat, const float* __restrict__ o_sum,
                  float* __restrict__ s_out) {
    __shared__ float s_l[NKM];     // [k][m], m rotated by k to avoid bank conflicts

    const int tid  = threadIdx.x;
    const int lane = tid & 31;
    const int w    = tid >> 5;
    const int b    = blockIdx.y;
    const int n0   = blockIdx.x * 128;

    for (int i = tid; i < NKM; i += 256) s_l[i] = 0.f;

    float o_reg[MCAP];
    #pragma unroll
    for (int m = 0; m < MCAP; ++m)
        o_reg[m] = o_sum[(size_t)b * NKM + lane * MCAP + m];

    __syncthreads();

    for (int t = 0; t < 16; ++t) {
        int n = n0 + w * 16 + t;
        const uint4* rp =
            reinterpret_cast<const uint4*>(hat + ((size_t)b * N_ + n) * NKM + lane * MCAP);
        uint4 q[8];
        #pragma unroll
        for (int i = 0; i < 8; ++i) q[i] = rp[i];

        // b_i[k,n] = sum_m o_sum[k,m] * hat[k,n,m]
        float logit = 0.f;
        #pragma unroll
        for (int i = 0; i < 8; ++i) {
            unsigned uu[4] = {q[i].x, q[i].y, q[i].z, q[i].w};
            #pragma unroll
            for (int d = 0; d < 4; ++d) {
                int m = i * 8 + d * 2;
                logit = fmaf(bflo(uu[d]), o_reg[m],     logit);
                logit = fmaf(bfhi(uu[d]), o_reg[m + 1], logit);
            }
        }
        // softmax over capsule axis == over the 32 lanes of this wave
        float mx = logit;
        #pragma unroll
        for (int off = 16; off; off >>= 1) mx = fmaxf(mx, __shfl_xor(mx, off, 32));
        float e  = __expf(logit - mx);
        float sm = e;
        #pragma unroll
        for (int off = 16; off; off >>= 1) sm += __shfl_xor(sm, off, 32);
        float c = e / sm;

        // s[k,m] += c * hat[k,n,m]  (LDS atomics, rotated index)
        #pragma unroll
        for (int i = 0; i < 8; ++i) {
            unsigned uu[4] = {q[i].x, q[i].y, q[i].z, q[i].w};
            #pragma unroll
            for (int d = 0; d < 4; ++d) {
                int m = i * 8 + d * 2;
                atomicAdd(&s_l[lane * MCAP + ((m     + lane) & 63)], c * bflo(uu[d]));
                atomicAdd(&s_l[lane * MCAP + ((m + 1 + lane) & 63)], c * bfhi(uu[d]));
            }
        }
    }
    __syncthreads();
    for (int idx = tid; idx < NKM; idx += 256) {
        int k = idx >> 6, m = idx & 63;
        atomicAdd(&s_out[(size_t)b * NKM + idx], s_l[k * MCAP + ((m + k) & 63)]);
    }
}

// ---------------------------------------------------------------- K2/K4/K6: squash
// Grid B*NCAP blocks of 64 threads; block = one (b,k) capsule vector.
// mode 0: out = squash(in * prescale)   mode 1: out += squash(in)   mode 2: out = squash(in)
__global__ __launch_bounds__(64)
void squash_kernel(const float* __restrict__ in, float* __restrict__ out,
                   float prescale, int mode) {
    __shared__ float red[2];
    int g = blockIdx.x * 64 + threadIdx.x;
    float v = in[g] * prescale;
    float sq = v * v;
    #pragma unroll
    for (int off = 16; off; off >>= 1) sq += __shfl_xor(sq, off, 32);
    if ((threadIdx.x & 31) == 0) red[threadIdx.x >> 5] = sq;
    __syncthreads();
    float scale = squash_scale(red[0] + red[1]);
    float o = v * scale;
    if (mode == 1) out[g] += o;
    else           out[g]  = o;
}

// ----------------------------------------------------------------
extern "C" void kernel_launch(void* const* d_in, const int* in_sizes, int n_in,
                              void* d_out, int out_size, void* d_ws, size_t ws_size,
                              hipStream_t stream) {
    (void)in_sizes; (void)n_in; (void)out_size; (void)ws_size;

    const float* x = (const float*)d_in[0];   // (32, 2048, 256) f32
    const float* W = (const float*)d_in[1];   // (256, 2048) f32
    float* out = (float*)d_out;               // (32, 32, 64) f32

    char* ws = (char*)d_ws;
    __bf16* hat  = (__bf16*)ws;                                   // 256 MB
    float*  S0   = (float*)(ws + (size_t)B_ * N_ * NKM * 2);      // 32x2048
    float*  s1   = S0 + (size_t)B_ * NKM;
    float*  s2   = s1 + (size_t)B_ * NKM;
    float*  osum = s2 + (size_t)B_ * NKM;

    // zero atomic targets (S0, s1, s2 are contiguous)
    zero_kernel<<<(3 * B_ * NKM + 255) / 256, 256, 0, stream>>>(S0, 3 * B_ * NKM);

    gemm_hat_kernel<<<dim3(16, 16, B_), 256, 0, stream>>>(x, W, hat, S0);

    squash_kernel<<<B_ * NCAP, 64, 0, stream>>>(S0, osum, 1.f / (float)NCAP, 0); // outputs0
    route_kernel<<<dim3(16, B_), 256, 0, stream>>>(hat, osum, s1);
    squash_kernel<<<B_ * NCAP, 64, 0, stream>>>(s1, osum, 1.f, 1);               // + outputs1
    route_kernel<<<dim3(16, B_), 256, 0, stream>>>(hat, osum, s2);
    squash_kernel<<<B_ * NCAP, 64, 0, stream>>>(s2, out, 1.f, 2);                // final
}
